// GraphTransformer_7413113553290
// MI455X (gfx1250) — matrix operand order repaired and checked by reading, not compiled, not run on hardware
//
#include <hip/hip_runtime.h>
#include <math.h>

typedef __attribute__((ext_vector_type(16))) _Float16 v16h;
typedef __attribute__((ext_vector_type(8)))  _Float16 v8h;
typedef __attribute__((ext_vector_type(4)))  _Float16 h4;
typedef __attribute__((ext_vector_type(8)))  float    v8f;

#define TM 128
#define TN 64
#define TK 32
#define LDA 40   // halves per sA row (32 + 8 pad), 80 B stride: 16-B aligned
#define LDB 40   // halves per sBt col

// ---------------------------------------------------------------------------
// f32 -> f16 bulk convert (vectorized), n4 = count/4
// ---------------------------------------------------------------------------
__global__ void conv_h(const float* __restrict__ X, _Float16* __restrict__ Y, int n4)
{
    int i = blockIdx.x * blockDim.x + threadIdx.x;
    if (i >= n4) return;
    float4 v = ((const float4*)X)[i];
    h4 hv; hv[0] = (_Float16)v.x; hv[1] = (_Float16)v.y;
           hv[2] = (_Float16)v.z; hv[3] = (_Float16)v.w;
    ((h4*)Y)[i] = hv;
}

// ---------------------------------------------------------------------------
// weight convert + transpose: W[K,Nout] f32 -> WT[Nout,K] f16 (coalesced writes)
// ---------------------------------------------------------------------------
__global__ void conv_wT(const float* __restrict__ W, _Float16* __restrict__ WT,
                        int logK, int Nout, int total)
{
    int o = blockIdx.x * blockDim.x + threadIdx.x;    // o = n*K + k
    if (o >= total) return;
    int K = 1 << logK;
    int n = o >> logK;
    int k = o & (K - 1);
    WT[o] = (_Float16)W[(size_t)k * Nout + n];
}

// ---------------------------------------------------------------------------
// Fused 4-way GEMM: C_z = A[M,K] x W_z[K,Nout] + b_z for z in {q,k,v,skip}.
// A is pre-converted f16; weights pre-converted+transposed f16 ([Nout,K]).
// LDS double-buffered; every WMMA fragment is two ds_load_b128.
// ---------------------------------------------------------------------------
__global__ __launch_bounds__(256) void gemm4_wmma(
    const _Float16* __restrict__ A, int M, int K, int Nout,
    const _Float16* __restrict__ WT0, const float* __restrict__ b0, float* __restrict__ C0,
    const _Float16* __restrict__ WT1, const float* __restrict__ b1, float* __restrict__ C1,
    const _Float16* __restrict__ WT2, const float* __restrict__ b2, float* __restrict__ C2,
    const _Float16* __restrict__ WT3, const float* __restrict__ b3, float* __restrict__ C3)
{
    // sA[buf][row][pos]: pos order = [K0..31] (contiguous)
    __shared__ _Float16 sA[2][TM][LDA];
    // sBt[buf][col][k]: lane half h reads 16 contiguous halves at k = h*16
    __shared__ _Float16 sBt[2][TN][LDB];

    const _Float16* WT; const float* bias; float* C;
    switch (blockIdx.z) {
        case 0:  WT = WT0; bias = b0; C = C0; break;
        case 1:  WT = WT1; bias = b1; C = C1; break;
        case 2:  WT = WT2; bias = b2; C = C2; break;
        default: WT = WT3; bias = b3; C = C3; break;
    }

    const int tid  = threadIdx.x;
    const int lane = tid & 31;
    const int wave = tid >> 5;
    const int half = lane >> 4;
    const int l16  = lane & 15;
    const int m0   = blockIdx.x * TM;
    const int n0   = blockIdx.y * TN;
    const int rbase = wave * 16;

    // A staging: 2 chunks of 8 halves per thread (128 rows x 4 groups)
    const int ar0 = tid >> 2;                // rows 0..63  (i=0), 64..127 (i=1)
    const int ag  = tid & 3;                 // 16-B group within row
    const int apb = ag * 8;  // contiguous pos {0,8,16,24}
    // B staging: 1 chunk of 8 halves per thread (64 cols x 4 groups)
    const int bcol = tid >> 2;
    const int bg   = tid & 3;

    const int KT = K / TK;
    v8f acc[4] = {};

    uint4 arg[2], brg;
    auto load_tiles = [&](int k0) {
        #pragma unroll
        for (int i = 0; i < 2; ++i) {
            int row = m0 + ar0 + i * 64;
            arg[i] = make_uint4(0u, 0u, 0u, 0u);
            if (row < M)
                arg[i] = *(const uint4*)(A + (size_t)row * K + k0 + ag * 8);
        }
        brg = *(const uint4*)(WT + (size_t)(n0 + bcol) * K + k0 + bg * 8);
    };
    auto store_tiles = [&](int buf) {
        #pragma unroll
        for (int i = 0; i < 2; ++i)
            *(uint4*)&sA[buf][ar0 + i * 64][apb] = arg[i];
        *(uint4*)&sBt[buf][bcol][bg * 8] = brg;
    };

    load_tiles(0);
    store_tiles(0);
    __syncthreads();

    for (int kt = 0; kt < KT; ++kt) {
        if (kt + 1 < KT) load_tiles((kt + 1) * TK);   // overlap with WMMA below

        const int cur = kt & 1;
        const v8h* pa = (const v8h*)&sA[cur][rbase + l16][half * 16];
        v8h a0 = pa[0], a1 = pa[1];
        v16h a = __builtin_shufflevector(a0, a1,
                 0,1,2,3,4,5,6,7,8,9,10,11,12,13,14,15);
        #pragma unroll
        for (int t = 0; t < 4; ++t) {
            const v8h* pb = (const v8h*)&sBt[cur][t * 16 + l16][half * 16];
            v8h b0v = pb[0], b1v = pb[1];
            v16h b = __builtin_shufflevector(b0v, b1v,
                     0,1,2,3,4,5,6,7,8,9,10,11,12,13,14,15);
            acc[t] = __builtin_amdgcn_wmma_f32_16x16x32_f16(
                false, a, false, b, (short)0, acc[t], false, false);
        }
        if (kt + 1 < KT) {
            store_tiles((kt + 1) & 1);
            __syncthreads();
        }
    }

    // epilogue: D layout is M = rbase + half*8 + r, N = t*16 + l16
    #pragma unroll
    for (int t = 0; t < 4; ++t) {
        int col = n0 + t * 16 + l16;
        float bv = bias[col];
        #pragma unroll
        for (int r = 0; r < 8; ++r) {
            int row = m0 + rbase + half * 8 + r;
            if (row < M)
                C[(size_t)row * Nout + col] = acc[t][r] + bv;
        }
    }
}

// ---------------------------------------------------------------------------
// Layer-3 tiny linear (Fin=256 -> 2) x4: q,k,v and skip (into agg).
// ---------------------------------------------------------------------------
__global__ __launch_bounds__(256) void lin_small4(
    const float* __restrict__ X, int M, int K,
    const float* __restrict__ Wq, const float* __restrict__ bq,
    const float* __restrict__ Wk, const float* __restrict__ bk,
    const float* __restrict__ Wv, const float* __restrict__ bv,
    const float* __restrict__ Ws, const float* __restrict__ bs,
    float* __restrict__ q, float* __restrict__ k,
    float* __restrict__ v, float* __restrict__ agg)
{
    int i = blockIdx.x * blockDim.x + threadIdx.x;
    if (i >= M) return;
    const float* xr = X + (size_t)i * K;
    float q0 = 0.f, q1 = 0.f, k0 = 0.f, k1 = 0.f, v0 = 0.f, v1 = 0.f, s0 = 0.f, s1 = 0.f;
    for (int c = 0; c < K; ++c) {
        float x = xr[c];
        q0 += x * Wq[c * 2 + 0]; q1 += x * Wq[c * 2 + 1];
        k0 += x * Wk[c * 2 + 0]; k1 += x * Wk[c * 2 + 1];
        v0 += x * Wv[c * 2 + 0]; v1 += x * Wv[c * 2 + 1];
        s0 += x * Ws[c * 2 + 0]; s1 += x * Ws[c * 2 + 1];
    }
    q[2 * i] = q0 + bq[0];   q[2 * i + 1] = q1 + bq[1];
    k[2 * i] = k0 + bk[0];   k[2 * i + 1] = k1 + bk[1];
    v[2 * i] = v0 + bv[0];   v[2 * i + 1] = v1 + bv[1];
    agg[2 * i] = s0 + bs[0]; agg[2 * i + 1] = s1 + bs[1];
}

// ---------------------------------------------------------------------------
// Edge-phase helpers
// ---------------------------------------------------------------------------
__device__ __forceinline__ void atomicMaxFloat(float* addr, float val) {
    if (!(val == val)) return;                    // skip NaN
    if (val >= 0.f) atomicMax((int*)addr, __float_as_int(val));
    else            atomicMin((unsigned int*)addr, __float_as_uint(val));
}

__global__ void init_md(float* __restrict__ m, float* __restrict__ den, int n) {
    int i = blockIdx.x * blockDim.x + threadIdx.x;
    if (i >= n) return;
    m[i]   = __int_as_float(0xFF800000);          // -inf
    den[i] = 0.f;
}

// pass 1: alpha[e,h] = <q[dst,h,:], k[src,h,:]> * scale ; m[dst,h] = max(...)
__global__ void edge_scores(
    const float* __restrict__ q, const float* __restrict__ k,
    const int* __restrict__ src, const int* __restrict__ dst,
    float* __restrict__ alpha, float* __restrict__ m,
    int E, int logH, int C, float scale)
{
    int H = 1 << logH;
    int idx = blockIdx.x * blockDim.x + threadIdx.x;
    if (idx >= E * H) return;
    int e = idx >> logH;
    int h = idx & (H - 1);
    int s = src[e], d = dst[e];
    const float* qp = q + (size_t)d * (H * C) + h * C;
    const float* kp = k + (size_t)s * (H * C) + h * C;
    float acc = 0.f;
    if ((C & 3) == 0) {
        for (int c = 0; c < C; c += 4) {
            float4 qv = *(const float4*)(qp + c);
            float4 kv = *(const float4*)(kp + c);
            acc += qv.x * kv.x + qv.y * kv.y + qv.z * kv.z + qv.w * kv.w;
        }
    } else {
        for (int c = 0; c < C; ++c) acc += qp[c] * kp[c];
    }
    acc *= scale;
    alpha[idx] = acc;
    atomicMaxFloat(m + (size_t)d * H + h, acc);
}

// pass 2: ex = exp(alpha - m[dst]); denom[dst] += ex (ex overwrites alpha)
__global__ void edge_softmax_num(
    const int* __restrict__ dst, const float* __restrict__ m,
    float* __restrict__ alpha, float* __restrict__ den, int E, int logH)
{
    int H = 1 << logH;
    int idx = blockIdx.x * blockDim.x + threadIdx.x;
    if (idx >= E * H) return;
    int e = idx >> logH;
    int h = idx & (H - 1);
    int d = dst[e];
    float ex = __expf(alpha[idx] - m[(size_t)d * H + h]);
    alpha[idx] = ex;
    atomicAdd(den + (size_t)d * H + h, ex);
}

// pass 3: agg[dst,f] += (ex/denom) * v[src,f], vec4 over F = H*C (F%4==0)
__global__ void edge_aggregate_vec4(
    const int* __restrict__ src, const int* __restrict__ dst,
    const float* __restrict__ ex, const float* __restrict__ den,
    const float* __restrict__ v, float* __restrict__ agg,
    int E, int logH, int logC)
{
    int H = 1 << logH;
    int logG = logH + logC - 2;                   // groups of 4 floats in F
    int idx = blockIdx.x * blockDim.x + threadIdx.x;
    if (idx >= (E << logG)) return;
    int e = idx >> logG;
    int g = idx & ((1 << logG) - 1);
    int f = g * 4;
    int h = f >> logC;
    int s = src[e], d = dst[e];
    int F = H << logC;
    float coeff = ex[(e << logH) + h] / (den[(size_t)d * H + h] + 1e-16f);
    float4 vv = *(const float4*)(v + (size_t)s * F + f);
    float* ap = agg + (size_t)d * F + f;
    atomicAdd(ap + 0, coeff * vv.x);
    atomicAdd(ap + 1, coeff * vv.y);
    atomicAdd(ap + 2, coeff * vv.z);
    atomicAdd(ap + 3, coeff * vv.w);
}

// pass 3 for layer 3 (F = 2)
__global__ void edge_aggregate_f2(
    const int* __restrict__ src, const int* __restrict__ dst,
    const float* __restrict__ ex, const float* __restrict__ den,
    const float* __restrict__ v, float* __restrict__ agg, int E)
{
    int e = blockIdx.x * blockDim.x + threadIdx.x;
    if (e >= E) return;
    int s = src[e], d = dst[e];
    float coeff = ex[e] / (den[d] + 1e-16f);
    atomicAdd(agg + 2 * d + 0, coeff * v[2 * s + 0]);
    atomicAdd(agg + 2 * d + 1, coeff * v[2 * s + 1]);
}

// ---------------------------------------------------------------------------
// Fused LayerNorm + ELU, one wave32 per row. OutT = f32 or f16.
// ---------------------------------------------------------------------------
template <typename OutT>
__global__ __launch_bounds__(256) void ln_elu(
    const float* __restrict__ X, const float* __restrict__ g,
    const float* __restrict__ b, OutT* __restrict__ Y, int rows, int F)
{
    int row  = blockIdx.x * (blockDim.x >> 5) + (threadIdx.x >> 5);
    int lane = threadIdx.x & 31;
    if (row >= rows) return;
    const float* xr = X + (size_t)row * F;
    float s = 0.f, s2 = 0.f;
    for (int c = lane * 4; c < F; c += 128) {
        float4 xv = *(const float4*)(xr + c);
        s  += xv.x + xv.y + xv.z + xv.w;
        s2 += xv.x * xv.x + xv.y * xv.y + xv.z * xv.z + xv.w * xv.w;
    }
    #pragma unroll
    for (int off = 16; off; off >>= 1) {
        s  += __shfl_xor(s,  off, 32);
        s2 += __shfl_xor(s2, off, 32);
    }
    float invF = 1.f / (float)F;
    float mu   = s * invF;
    float var  = s2 * invF - mu * mu;
    float inv  = rsqrtf(var + 1e-5f);
    OutT* yr   = Y + (size_t)row * F;
    for (int c = lane * 4; c < F; c += 128) {
        float4 xv = *(const float4*)(xr + c);
        float o0 = (xv.x - mu) * inv * g[c + 0] + b[c + 0];
        float o1 = (xv.y - mu) * inv * g[c + 1] + b[c + 1];
        float o2 = (xv.z - mu) * inv * g[c + 2] + b[c + 2];
        float o3 = (xv.w - mu) * inv * g[c + 3] + b[c + 3];
        yr[c + 0] = (OutT)(o0 > 0.f ? o0 : expm1f(o0));
        yr[c + 1] = (OutT)(o1 > 0.f ? o1 : expm1f(o1));
        yr[c + 2] = (OutT)(o2 > 0.f ? o2 : expm1f(o2));
        yr[c + 3] = (OutT)(o3 > 0.f ? o3 : expm1f(o3));
    }
}

// ---------------------------------------------------------------------------
// Final log_softmax over 2 classes
// ---------------------------------------------------------------------------
__global__ void final_log_softmax(const float* __restrict__ agg,
                                  float* __restrict__ out, int n)
{
    int i = blockIdx.x * blockDim.x + threadIdx.x;
    if (i >= n) return;
    float h0 = agg[2 * i], h1 = agg[2 * i + 1];
    float mx = fmaxf(h0, h1);
    float lse = mx + logf(__expf(h0 - mx) + __expf(h1 - mx));
    out[2 * i] = h0 - lse;
    out[2 * i + 1] = h1 - lse;
}

// ---------------------------------------------------------------------------
extern "C" void kernel_launch(void* const* d_in, const int* in_sizes, int n_in,
                              void* d_out, int out_size, void* d_ws, size_t ws_size,
                              hipStream_t stream)
{
    const int F_IN = 128;
    const int N = in_sizes[0] / F_IN;
    const int E = in_sizes[1] / 2;

    const float* x   = (const float*)d_in[0];
    const int*   ei  = (const int*)d_in[1];
    const int*   src = ei;
    const int*   dst = ei + E;

    const float* Wq1 = (const float*)d_in[2],  *bq1 = (const float*)d_in[3];
    const float* Wk1 = (const float*)d_in[4],  *bk1 = (const float*)d_in[5];
    const float* Wv1 = (const float*)d_in[6],  *bv1 = (const float*)d_in[7];
    const float* Ws1 = (const float*)d_in[8],  *bs1 = (const float*)d_in[9];
    const float* Wq2 = (const float*)d_in[10], *bq2 = (const float*)d_in[11];
    const float* Wk2 = (const float*)d_in[12], *bk2 = (const float*)d_in[13];
    const float* Wv2 = (const float*)d_in[14], *bv2 = (const float*)d_in[15];
    const float* Ws2 = (const float*)d_in[16], *bs2 = (const float*)d_in[17];
    const float* Wq3 = (const float*)d_in[18], *bq3 = (const float*)d_in[19];
    const float* Wk3 = (const float*)d_in[20], *bk3 = (const float*)d_in[21];
    const float* Wv3 = (const float*)d_in[22], *bv3 = (const float*)d_in[23];
    const float* Ws3 = (const float*)d_in[24], *bs3 = (const float*)d_in[25];
    const float* g1  = (const float*)d_in[26], *be1 = (const float*)d_in[27];
    const float* g2  = (const float*)d_in[28], *be2 = (const float*)d_in[29];

    // workspace carve-out
    char* ws = (char*)d_ws;
    const size_t NB = (size_t)N * 512 * sizeof(float);     // 102.4 MB
    float* Q    = (float*)(ws + 0 * NB);
    float* Kb   = (float*)(ws + 1 * NB);
    float* V    = (float*)(ws + 2 * NB);
    float* AGG  = (float*)(ws + 3 * NB);
    float* Hbuf = (float*)(ws + 4 * NB);
    float* ALPHA = (float*)(ws + 5 * NB);
    float* Mb    = ALPHA + (size_t)E * 4;
    float* Db    = Mb + (size_t)N * 4;
    _Float16* Ah = (_Float16*)(Db + (size_t)N * 4);        // N*512 halves
    _Float16* WT[4];
    {
        _Float16* wt = Ah + (size_t)N * 512;
        for (int z = 0; z < 4; ++z) { WT[z] = wt; wt += 512 * 256; }
    }

    const int mb = (N + TM - 1) / TM;

    // =================== Layer 1: 128 -> 512, H=4, C=128 ===================
    conv_h<<<(N * 128 / 4 + 255) / 256, 256, 0, stream>>>(x, Ah, N * 128 / 4);
    conv_wT<<<(128 * 512 + 255) / 256, 256, 0, stream>>>(Wq1, WT[0], 7, 512, 128 * 512);
    conv_wT<<<(128 * 512 + 255) / 256, 256, 0, stream>>>(Wk1, WT[1], 7, 512, 128 * 512);
    conv_wT<<<(128 * 512 + 255) / 256, 256, 0, stream>>>(Wv1, WT[2], 7, 512, 128 * 512);
    conv_wT<<<(128 * 512 + 255) / 256, 256, 0, stream>>>(Ws1, WT[3], 7, 512, 128 * 512);
    gemm4_wmma<<<dim3(mb, 512 / TN, 4), 256, 0, stream>>>(
        Ah, N, 128, 512, WT[0], bq1, Q, WT[1], bk1, Kb, WT[2], bv1, V, WT[3], bs1, AGG);
    init_md<<<(N * 4 + 255) / 256, 256, 0, stream>>>(Mb, Db, N * 4);
    edge_scores<<<(E * 4 + 255) / 256, 256, 0, stream>>>(
        Q, Kb, src, dst, ALPHA, Mb, E, 2, 128, rsqrtf(128.f));
    edge_softmax_num<<<(E * 4 + 255) / 256, 256, 0, stream>>>(dst, Mb, ALPHA, Db, E, 2);
    edge_aggregate_vec4<<<((E << 7) + 255) / 256, 256, 0, stream>>>(
        src, dst, ALPHA, Db, V, AGG, E, 2, 7);
    // LN+ELU -> f16 activations for the next GEMM
    ln_elu<_Float16><<<(N + 7) / 8, 256, 0, stream>>>(AGG, g1, be1, Ah, N, 512);

    // =================== Layer 2: 512 -> 256, H=4, C=64 ====================
    conv_wT<<<(512 * 256 + 255) / 256, 256, 0, stream>>>(Wq2, WT[0], 9, 256, 512 * 256);
    conv_wT<<<(512 * 256 + 255) / 256, 256, 0, stream>>>(Wk2, WT[1], 9, 256, 512 * 256);
    conv_wT<<<(512 * 256 + 255) / 256, 256, 0, stream>>>(Wv2, WT[2], 9, 256, 512 * 256);
    conv_wT<<<(512 * 256 + 255) / 256, 256, 0, stream>>>(Ws2, WT[3], 9, 256, 512 * 256);
    gemm4_wmma<<<dim3(mb, 256 / TN, 4), 256, 0, stream>>>(
        Ah, N, 512, 256, WT[0], bq2, Q, WT[1], bk2, Kb, WT[2], bv2, V, WT[3], bs2, AGG);
    init_md<<<(N * 4 + 255) / 256, 256, 0, stream>>>(Mb, Db, N * 4);
    edge_scores<<<(E * 4 + 255) / 256, 256, 0, stream>>>(
        Q, Kb, src, dst, ALPHA, Mb, E, 2, 64, rsqrtf(64.f));
    edge_softmax_num<<<(E * 4 + 255) / 256, 256, 0, stream>>>(dst, Mb, ALPHA, Db, E, 2);
    edge_aggregate_vec4<<<((E << 6) + 255) / 256, 256, 0, stream>>>(
        src, dst, ALPHA, Db, V, AGG, E, 2, 6);
    ln_elu<float><<<(N + 7) / 8, 256, 0, stream>>>(AGG, g2, be2, Hbuf, N, 256);

    // =================== Layer 3: 256 -> 2, H=1, C=2 =======================
    lin_small4<<<(N + 255) / 256, 256, 0, stream>>>(
        Hbuf, N, 256, Wq3, bq3, Wk3, bk3, Wv3, bv3, Ws3, bs3, Q, Kb, V, AGG);
    init_md<<<(N + 255) / 256, 256, 0, stream>>>(Mb, Db, N);
    edge_scores<<<(E + 255) / 256, 256, 0, stream>>>(
        Q, Kb, src, dst, ALPHA, Mb, E, 0, 2, rsqrtf(2.f));
    edge_softmax_num<<<(E + 255) / 256, 256, 0, stream>>>(dst, Mb, ALPHA, Db, E, 0);
    edge_aggregate_f2<<<(E + 255) / 256, 256, 0, stream>>>(src, dst, ALPHA, Db, V, AGG, E);
    final_log_softmax<<<(N + 255) / 256, 256, 0, stream>>>(AGG, (float*)d_out, N);
}
